// TransformerHATLayer_3229815407007
// MI455X (gfx1250) — matrix operand order repaired and checked by reading, not compiled, not run on hardware
//
#include <hip/hip_runtime.h>

#define XS  132   // LDS row stride (floats) for f32 16x128 tiles
#define XBS 136   // LDS row stride (halves) for bf16 16x128 tiles (16B-aligned rows)
#define PBS 24    // LDS row stride (halves) for 16x16 bf16 prob tiles

typedef __bf16 bf16;
typedef __attribute__((ext_vector_type(16))) __bf16 v16bf;
typedef __attribute__((ext_vector_type(8)))  __bf16 v8bf;
typedef __attribute__((ext_vector_type(8)))  float  v8f;

__device__ __forceinline__ v8f wmma_bf16(v16bf a, v16bf b, v8f c) {
  return __builtin_amdgcn_wmma_f32_16x16x32_bf16(false, a, false, b, (short)0, c, false, false);
}

// A fragment from bf16 LDS tile (row-major, stride in halves), K-step k0.
// Contiguous 16-bit A layout: lane L<16: row L, elems 0..15 = K k0..k0+15;
// lanes 16..31: same rows, K k0+16..k0+31. Two 16B ds loads, no VALU cvt.
__device__ __forceinline__ v16bf load_afrag_b(const bf16* A, int stride, int k0, int lane) {
  int r  = lane & 15;
  int kb = k0 + ((lane >> 4) << 4);
  const v8bf* p0 = reinterpret_cast<const v8bf*>(A + r * stride + kb);
  const v8bf* p1 = reinterpret_cast<const v8bf*>(A + r * stride + kb + 8);
  v8bf x = *p0, y = *p1;
  return __builtin_shufflevector(x, y, 0,1,2,3,4,5,6,7,8,9,10,11,12,13,14,15);
}

// C = [Res +] act( A(16xK bf16 LDS) @ WT^T + bias ); static config -> branch-free epilogue.
// WT: bf16, N x KPAD row-major (pre-transposed weight) -> contiguous 32B B-fragment loads.
// MODE 0: C = A@W + bias ; MODE 1: C = Res + relu(A@W + bias)
template<int K, int KPAD, int MODE, bool HASF, bool HASB>
__device__ __forceinline__ void gemm16(const bf16* Ab, int astride, const bf16* WT,
                                       const float* bias, float* Cf, bf16* Cb,
                                       const float* Res, int lane, int wave) {
  int n0a = wave * 32;
  int n0b = n0a + 16;
  v8f acc0 = {0.f,0.f,0.f,0.f,0.f,0.f,0.f,0.f};
  v8f acc1 = acc0;
#pragma unroll
  for (int k0 = 0; k0 < K; k0 += 32) {
    v16bf a = load_afrag_b(Ab, astride, k0, lane);
    int kk = k0 + ((lane >> 4) << 4);
    v16bf b0 = *reinterpret_cast<const v16bf*>(WT + (size_t)(n0a + (lane & 15)) * KPAD + kk);
    v16bf b1 = *reinterpret_cast<const v16bf*>(WT + (size_t)(n0b + (lane & 15)) * KPAD + kk);
    acc0 = wmma_bf16(a, b0, acc0);
    acc1 = wmma_bf16(a, b1, acc1);
  }
  int rb = (lane >> 4) << 3;
#pragma unroll
  for (int t = 0; t < 2; ++t) {
    v8f acc = t ? acc1 : acc0;
    int col = (t ? n0b : n0a) + (lane & 15);
    float bv = bias[col];
#pragma unroll
    for (int r = 0; r < 8; ++r) {
      float x = acc[r] + bv;
      if (MODE == 1) x = Res[(rb + r) * XS + col] + fmaxf(x, 0.f);
      if (HASF) Cf[(rb + r) * XS + col] = x;
      if (HASB) Cb[(rb + r) * XBS + col] = (bf16)x;
    }
  }
}

// One attention head (wave==head): S = Qh Kh^T /sqrt(128); softmax cols<nk; O = Qp + A Vh.
// Qp dual (f32 residual + bf16 A-source); Kp/Vp bf16-only; P written/read as bf16.
__device__ __forceinline__ void attention_head(const float* Qpf, const bf16* Qpb,
                                               const bf16* Kb, const bf16* Vb,
                                               float* Of, bf16* Ob, bf16* Phb,
                                               int nk, int lane, int head) {
  int c0 = head * 32;
  v16bf a = load_afrag_b(Qpb, XBS, c0, lane);
  v16bf b;
  {
    int n  = lane & 15;
    int kk = (lane >> 4) << 4;
    const v8bf* p = reinterpret_cast<const v8bf*>(Kb + n * XBS + c0 + kk);
    v8bf x = p[0], y = p[1];
    b = __builtin_shufflevector(x, y, 0,1,2,3,4,5,6,7,8,9,10,11,12,13,14,15);
  }
  v8f zero = {0.f,0.f,0.f,0.f,0.f,0.f,0.f,0.f};
  v8f s = wmma_bf16(a, b, zero);
  const float scale = 0.08838834764831845f; // 1/sqrt(128)
  bool mask = (lane & 15) >= nk;
#pragma unroll
  for (int r = 0; r < 8; ++r) s[r] = mask ? -1e30f : s[r] * scale;
  // softmax across the 16-lane column group (xor<=8 stays within group of 16)
#pragma unroll
  for (int r = 0; r < 8; ++r) {
    float m = s[r];
    m = fmaxf(m, __shfl_xor(m, 1, 32));
    m = fmaxf(m, __shfl_xor(m, 2, 32));
    m = fmaxf(m, __shfl_xor(m, 4, 32));
    m = fmaxf(m, __shfl_xor(m, 8, 32));
    float p = __expf(s[r] - m);
    float sum = p;
    sum += __shfl_xor(sum, 1, 32);
    sum += __shfl_xor(sum, 2, 32);
    sum += __shfl_xor(sum, 4, 32);
    sum += __shfl_xor(sum, 8, 32);
    s[r] = p / sum;
  }
  {
    int rb = (lane >> 4) << 3, c = lane & 15;
#pragma unroll
    for (int r = 0; r < 8; ++r) Phb[(rb + r) * PBS + c] = (bf16)s[r];
  }
  __syncthreads();
  v16bf pa;
  {
    int r  = lane & 15;
    const v8bf* p = reinterpret_cast<const v8bf*>(Phb + r * PBS);
    v8bf x = p[0], y = p[1];
    v8bf z;
#pragma unroll
    for (int e = 0; e < 8; ++e) z[e] = (bf16)0.0f;
    pa = (lane < 16) ? __builtin_shufflevector(x, y, 0,1,2,3,4,5,6,7,8,9,10,11,12,13,14,15)
                     : __builtin_shufflevector(z, z, 0,1,2,3,4,5,6,7,8,9,10,11,12,13,14,15);
  }
#pragma unroll
  for (int t = 0; t < 2; ++t) {
    int n0 = t * 16;
    v16bf vb;
    if (lane < 16) {
      const bf16* p = Vb + c0 + n0 + lane;  // column reads, rows e=0..15
#pragma unroll
      for (int e = 0; e < 16; ++e) vb[e] = p[e * XBS];
    } else {
#pragma unroll
      for (int e = 0; e < 16; ++e) vb[e] = (bf16)0.0f;  // K pad 16..31
    }
    int col = c0 + n0 + (lane & 15);
    int rb  = (lane >> 4) << 3;
    v8f c;
#pragma unroll
    for (int r = 0; r < 8; ++r) c[r] = Qpf[(rb + r) * XS + col];  // residual preload
    v8f o = wmma_bf16(pa, vb, c);
#pragma unroll
    for (int r = 0; r < 8; ++r) {
      Of[(rb + r) * XS + col]  = o[r];
      Ob[(rb + r) * XBS + col] = (bf16)o[r];
    }
  }
}

struct MabW {
  const bf16 *qW, *kW, *vW, *oW;
  const float *qb, *kb, *vb, *ob;
};

__device__ __forceinline__ void mab_run(const bf16* Qsrc, const bf16* Ksrc, int nk,
                                        const MabW& w,
                                        float* S1f, bf16* S1b, bf16* Kb, bf16* Vb,
                                        float* S4f, bf16* S4b, bf16* Pb,
                                        bf16* Outb, int lane, int wave) {
  gemm16<128,128,0,true, true >(Qsrc, XBS, w.qW, w.qb, S1f, S1b, nullptr, lane, wave); // Qp dual
  gemm16<128,128,0,false,true >(Ksrc, XBS, w.kW, w.kb, nullptr, Kb, nullptr, lane, wave); // Kp
  gemm16<128,128,0,false,true >(Ksrc, XBS, w.vW, w.vb, nullptr, Vb, nullptr, lane, wave); // Vp
  __syncthreads();
  attention_head(S1f, S1b, Kb, Vb, S4f, S4b, Pb + wave * 16 * PBS, nk, lane, wave);
  __syncthreads();
  gemm16<128,128,1,false,true >(S4b, XBS, w.oW, w.ob, nullptr, Outb, S4f, lane, wave);  // out
  __syncthreads();
}

struct EdgeArgs {
  const float* vfeat; const float* efeat; const float* weight;
  const int* edge2node;
  const bf16* peWT; const float* peb;
  const float* I0; const float* I1;
  MabW m[5];                       // isab0.mab0, isab0.mab1, isab1.mab0, isab1.mab1, dec_mab
  const bf16* dlWT; const float* dlb;
  const bf16* keWT; const float* keb;
  const bf16* veWT; const float* veb;
  float* efeat_out; float* k_e; float* vv_e;
};

__global__ void __launch_bounds__(128) edge_kernel(EdgeArgs args) {
  __shared__ float S1f[16 * XS], S4f[16 * XS];
  __shared__ bf16  Xb[16 * XBS], Qbb[16 * XBS], Hbb[16 * XBS];
  __shared__ bf16  Kb[16 * XBS], Vb[16 * XBS], S1b[16 * XBS], S4b[16 * XBS];
  __shared__ bf16  Pb[4 * 16 * PBS];
  int e = blockIdx.x;
  int tid = threadIdx.x, lane = tid & 31, wave = tid >> 5;

  if (tid == 0) __builtin_prefetch(args.efeat + (size_t)e * 128, 0, 3);  // consumed late

  // 1) pe_v: weight tile (16x48, zero-pad K to 64) staged bf16 in Xb; S1f = tile @ peW + peb
  for (int i = tid; i < 16 * 64; i += 128) {
    int r = i >> 6, c = i & 63;
    Xb[r * XBS + c] = (bf16)((c < 48) ? args.weight[((size_t)e * 16 + r) * 48 + c] : 0.f);
  }
  __syncthreads();
  gemm16<64,64,0,true,false>(Xb, XBS, args.peWT, args.peb, S1f, nullptr, nullptr, lane, wave);
  __syncthreads();
  // 2) X = vfeat[edge2node] + pe  (bf16-only, A-operand)
  for (int i = tid; i < 16 * 128; i += 128) {
    int r = i >> 7, c = i & 127;
    int node = args.edge2node[(size_t)e * 16 + r];
    Xb[r * XBS + c] = (bf16)(args.vfeat[(size_t)node * 128 + c] + S1f[r * XS + c]);
  }
  __syncthreads();
  // 3) two ISAB layers
  const float* Is[2] = {args.I0, args.I1};
#pragma unroll
  for (int L = 0; L < 2; ++L) {
    for (int i = tid; i < 16 * 128; i += 128) {
      int r = i >> 7, c = i & 127;
      Qbb[r * XBS + c] = (bf16)((r < 4) ? Is[L][r * 128 + c] : 0.f);
    }
    __syncthreads();
    mab_run(Qbb, Xb, 16, args.m[L*2+0], S1f,S1b,Kb,Vb,S4f,S4b,Pb, Hbb, lane, wave); // H=MAB(I,X)
    mab_run(Xb, Hbb, 4, args.m[L*2+1], S1f,S1b,Kb,Vb,S4f,S4b,Pb, Xb,  lane, wave); // X=MAB(X,H)
  }
  // 4) decoder MAB: Q = efeat[e] (row 0)
  for (int i = tid; i < 16 * 128; i += 128) {
    int r = i >> 7, c = i & 127;
    Qbb[r * XBS + c] = (bf16)((r == 0) ? args.efeat[(size_t)e * 128 + c] : 0.f);
  }
  __syncthreads();
  mab_run(Qbb, Xb, 16, args.m[4], S1f,S1b,Kb,Vb,S4f,S4b,Pb, Hbb, lane, wave);
  // 5) efeat_out = o @ dec_lin + b ; then k_e, vv_e projections of efeat_out
  gemm16<128,128,0,true,true>(Hbb, XBS, args.dlWT, args.dlb, S1f, Qbb, nullptr, lane, wave);
  __syncthreads();
  args.efeat_out[(size_t)e * 128 + tid] = S1f[tid];                    // row 0
  gemm16<128,128,0,true,false>(Qbb, XBS, args.keWT, args.keb, S4f, nullptr, nullptr, lane, wave);
  __syncthreads();
  args.k_e[(size_t)e * 128 + tid] = S4f[tid];                          // row 0
  __syncthreads();
  gemm16<128,128,0,true,false>(Qbb, XBS, args.veWT, args.veb, S4f, nullptr, nullptr, lane, wave);
  __syncthreads();
  args.vv_e[(size_t)e * 128 + tid] = S4f[tid];                         // row 0
}

struct NodeArgs {
  const float* vfeat; const int* node2edge;
  const bf16* qvWT; const float* qvb;
  const float* k_e; const float* vv_e;
  float* vfeat_out;
};

__global__ void __launch_bounds__(128) node_kernel(NodeArgs args) {
  __shared__ bf16  Xlb[16 * XBS];
  __shared__ float Ql[16 * XS];
  __shared__ float aP[16][8];
  __shared__ int   eI[16][8];
  int tid = threadIdx.x, lane = tid & 31, wave = tid >> 5;
  int base = blockIdx.x * 16;
  for (int i = tid; i < 16 * 128; i += 128) {
    int r = i >> 7, c = i & 127;
    Xlb[r * XBS + c] = (bf16)args.vfeat[(size_t)(base + r) * 128 + c];
  }
  __syncthreads();
  gemm16<128,128,0,true,false>(Xlb, XBS, args.qvWT, args.qvb, Ql, nullptr, nullptr, lane, wave);
  __syncthreads();
  int nl = tid >> 3, d = tid & 7;
  int e = args.node2edge[(size_t)(base + nl) * 8 + d];
  eI[nl][d] = e;
  const float* kp = args.k_e + (size_t)e * 128;
  float s = 0.f;
  for (int c = 0; c < 128; ++c) s += kp[c] * Ql[nl * XS + c];
  s = (s > 0.f ? s : 0.01f * s) * 0.08838834764831845f;  // leaky_relu then /sqrt(128)
  float m = s;
  m = fmaxf(m, __shfl_xor(m, 1, 32));
  m = fmaxf(m, __shfl_xor(m, 2, 32));
  m = fmaxf(m, __shfl_xor(m, 4, 32));
  float p = __expf(s - m);
  float sum = p;
  sum += __shfl_xor(sum, 1, 32);
  sum += __shfl_xor(sum, 2, 32);
  sum += __shfl_xor(sum, 4, 32);
  aP[nl][d] = p / sum;
  __syncthreads();
  for (int k = 0; k < 16; ++k) {
    float acc = 0.f;
#pragma unroll
    for (int dd = 0; dd < 8; ++dd)
      acc += aP[k][dd] * args.vv_e[(size_t)eI[k][dd] * 128 + tid];
    args.vfeat_out[(size_t)(base + k) * 128 + tid] = fmaxf(acc, 0.f);
  }
}

// Weight pre-transpose + f32->bf16: WT[n*Kpad + k] = W[k*N + n] (zero-pad k>=K)
__global__ void prep_wt(const float* W, bf16* WT, int K, int N, int Kpad) {
  int i = blockIdx.x * blockDim.x + threadIdx.x;
  if (i >= N * Kpad) return;
  int n = i / Kpad, k = i % Kpad;
  WT[i] = (k < K) ? (bf16)W[(size_t)k * N + n] : (bf16)0.0f;
}

extern "C" void kernel_launch(void* const* d_in, const int* in_sizes, int n_in,
                              void* d_out, int out_size, void* d_ws, size_t ws_size,
                              hipStream_t stream) {
  (void)in_sizes; (void)n_in; (void)out_size; (void)ws_size;
  const int N = 50000, E = 25000;
  const float* vfeat  = (const float*)d_in[0];
  const float* efeat  = (const float*)d_in[1];
  const float* weight = (const float*)d_in[2];
  const int* edge2node = (const int*)d_in[3];
  const int* node2edge = (const int*)d_in[4];
  // params flattened in JAX pytree order (dict keys sorted; 'W' < 'b')
  enum {
    P_DL_W = 5, P_DL_B,
    P_DEC_K_W, P_DEC_K_B, P_DEC_O_W, P_DEC_O_B, P_DEC_Q_W, P_DEC_Q_B, P_DEC_V_W, P_DEC_V_B,
    P_I0,
    P_00_K_W, P_00_K_B, P_00_O_W, P_00_O_B, P_00_Q_W, P_00_Q_B, P_00_V_W, P_00_V_B,
    P_01_K_W, P_01_K_B, P_01_O_W, P_01_O_B, P_01_Q_W, P_01_Q_B, P_01_V_W, P_01_V_B,
    P_I1,
    P_10_K_W, P_10_K_B, P_10_O_W, P_10_O_B, P_10_Q_W, P_10_Q_B, P_10_V_W, P_10_V_B,
    P_11_K_W, P_11_K_B, P_11_O_W, P_11_O_B, P_11_Q_W, P_11_Q_B, P_11_V_W, P_11_V_B,
    P_KE_W, P_KE_B, P_PE_W, P_PE_B, P_QV_W, P_QV_B, P_VE_W, P_VE_B
  };

  char* ws = (char*)d_ws;
  size_t off = 0;
  auto alloc = [&](size_t bytes) -> void* {
    void* p = ws + off;
    off = (off + bytes + 255) & ~(size_t)255;
    return p;
  };
  bf16* peWT = (bf16*)alloc(128 * 64 * sizeof(bf16));
  bf16* WT[24];
  for (int i = 0; i < 24; ++i) WT[i] = (bf16*)alloc(128 * 128 * sizeof(bf16));
  float* k_e  = (float*)alloc((size_t)E * 128 * sizeof(float));
  float* vv_e = (float*)alloc((size_t)E * 128 * sizeof(float));

  auto prep = [&](int pidx, bf16* dst, int K, int Np, int Kpad) {
    int total = Np * Kpad;
    prep_wt<<<(total + 255) / 256, 256, 0, stream>>>((const float*)d_in[pidx], dst, K, Np, Kpad);
  };
  prep(P_PE_W, peWT, 48, 128, 64);
  const int qkvoW[5][4] = {
    {P_00_Q_W, P_00_K_W, P_00_V_W, P_00_O_W},
    {P_01_Q_W, P_01_K_W, P_01_V_W, P_01_O_W},
    {P_10_Q_W, P_10_K_W, P_10_V_W, P_10_O_W},
    {P_11_Q_W, P_11_K_W, P_11_V_W, P_11_O_W},
    {P_DEC_Q_W, P_DEC_K_W, P_DEC_V_W, P_DEC_O_W}};
  const int qkvoB[5][4] = {
    {P_00_Q_B, P_00_K_B, P_00_V_B, P_00_O_B},
    {P_01_Q_B, P_01_K_B, P_01_V_B, P_01_O_B},
    {P_10_Q_B, P_10_K_B, P_10_V_B, P_10_O_B},
    {P_11_Q_B, P_11_K_B, P_11_V_B, P_11_O_B},
    {P_DEC_Q_B, P_DEC_K_B, P_DEC_V_B, P_DEC_O_B}};
  for (int mi = 0; mi < 5; ++mi)
    for (int j = 0; j < 4; ++j)
      prep(qkvoW[mi][j], WT[mi * 4 + j], 128, 128, 128);
  prep(P_DL_W, WT[20], 128, 128, 128);
  prep(P_KE_W, WT[21], 128, 128, 128);
  prep(P_VE_W, WT[22], 128, 128, 128);
  prep(P_QV_W, WT[23], 128, 128, 128);

  float* out = (float*)d_out;
  float* vfeat_out = out;                       // first output (N*128)
  float* efeat_out = out + (size_t)N * 128;     // second output (E*128)

  EdgeArgs ea;
  ea.vfeat = vfeat; ea.efeat = efeat; ea.weight = weight; ea.edge2node = edge2node;
  ea.peWT = peWT; ea.peb = (const float*)d_in[P_PE_B];
  ea.I0 = (const float*)d_in[P_I0]; ea.I1 = (const float*)d_in[P_I1];
  for (int mi = 0; mi < 5; ++mi) {
    ea.m[mi].qW = WT[mi * 4 + 0]; ea.m[mi].kW = WT[mi * 4 + 1];
    ea.m[mi].vW = WT[mi * 4 + 2]; ea.m[mi].oW = WT[mi * 4 + 3];
    ea.m[mi].qb = (const float*)d_in[qkvoB[mi][0]];
    ea.m[mi].kb = (const float*)d_in[qkvoB[mi][1]];
    ea.m[mi].vb = (const float*)d_in[qkvoB[mi][2]];
    ea.m[mi].ob = (const float*)d_in[qkvoB[mi][3]];
  }
  ea.dlWT = WT[20]; ea.dlb = (const float*)d_in[P_DL_B];
  ea.keWT = WT[21]; ea.keb = (const float*)d_in[P_KE_B];
  ea.veWT = WT[22]; ea.veb = (const float*)d_in[P_VE_B];
  ea.efeat_out = efeat_out; ea.k_e = k_e; ea.vv_e = vv_e;

  edge_kernel<<<E, 128, 0, stream>>>(ea);

  NodeArgs na;
  na.vfeat = vfeat; na.node2edge = node2edge;
  na.qvWT = WT[23]; na.qvb = (const float*)d_in[P_QV_B];
  na.k_e = k_e; na.vv_e = vv_e; na.vfeat_out = vfeat_out;
  node_kernel<<<N / 16, 128, 0, stream>>>(na);
}
